// CRFDecoder_33217277067524
// MI455X (gfx1250) — matrix operand order repaired and checked by reading, not compiled, not run on hardware
//
#include <hip/hip_runtime.h>

// Problem constants from the reference
#define B_   64
#define S_   512
#define H_   768
#define T_   50
#define NPAD 64                  // pad tag dim 50 -> 64 (4 WMMA N-tiles)
#define M_   (B_ * S_)           // 32768 rows of emissions
#define KBLK (H_ / 32)           // 24 k-blocks of 32

// workspace layout (bytes)
#define EM_BYTES   ((size_t)M_ * NPAD * 4)            // 8.39 MB fp32 emissions
#define WPK_ELEMS  ((size_t)KBLK * NPAD * 32)         // 49152 bf16 per plane
#define WPK_BYTES  (WPK_ELEMS * 2)                    // 96 KB per plane

typedef __attribute__((ext_vector_type(4)))  float   f4;
typedef __attribute__((ext_vector_type(8)))  float   v8f;
typedef __attribute__((ext_vector_type(16))) __bf16  v16bf;

// ---------------------------------------------------------------------------
// Prep kernel: pack W (H x T, fp32) into WMMA B-fragment order, split bf16
// hi/lo, padded to NPAD columns with zeros; also pad the bias to NPAD.
// Wpack[kblk][col][kk] holds K = kblk*32 + kk of column col. A lane's whole
// 16-element fragment is then 32 contiguous bytes.
// ---------------------------------------------------------------------------
__global__ __launch_bounds__(256)
void pack_w_kernel(const float* __restrict__ Wm, const float* __restrict__ bias,
                   __bf16* __restrict__ Whi, __bf16* __restrict__ Wlo,
                   float* __restrict__ biaspad)
{
    int idx = blockIdx.x * 256 + threadIdx.x;   // over KBLK*NPAD*32 = 49152
    int kblk = idx >> 11;                       // / (NPAD*32)
    int rem  = idx & 2047;
    int col  = rem >> 5;
    int kk   = rem & 31;
    int k    = kblk * 32 + kk;
    float v  = (col < T_) ? Wm[(size_t)k * T_ + col] : 0.0f;
    __bf16 h = (__bf16)v;
    Whi[idx] = h;
    Wlo[idx] = (__bf16)(v - (float)h);
    if (blockIdx.x == 0 && threadIdx.x < NPAD)
        biaspad[threadIdx.x] = (threadIdx.x < T_) ? bias[threadIdx.x] : 0.0f;
}

// ---------------------------------------------------------------------------
// Kernel 1: emissions = hidden @ W + b   (bf16x3 split-precision WMMA GEMM)
// 256 blocks x 256 threads (8 waves). Each wave owns a 16(M) x 64(N) stripe:
// A fragment loaded/converted once -> 12 WMMAs per k-iteration.
// ---------------------------------------------------------------------------
__global__ __launch_bounds__(256)
void emis_gemm_kernel(const float*  __restrict__ A,      // [M_, H_]
                      const __bf16* __restrict__ Whi,    // packed B hi plane
                      const __bf16* __restrict__ Wlo,    // packed B lo plane
                      const float*  __restrict__ biaspad,// [NPAD]
                      float* __restrict__ Em)            // [M_, NPAD]
{
    const int lane  = threadIdx.x & 31;
    const int wave  = threadIdx.x >> 5;
    const int mtile = blockIdx.x * 8 + wave;
    const int mbase = mtile * 16;
    const int ln15  = lane & 15;
    const int row   = mbase + ln15;
    const int kOffA = (lane < 16) ? 0 : 16;   // A frag: low lanes K 0..15
    const int kOffB = (lane < 16) ? 0 : 16;   // B frag: low lanes K 0..15

    const float* arow = A + (size_t)row * H_;

    // bias for this lane's 4 columns (no conditionals: biaspad is pre-zeroed)
    float bb[4];
#pragma unroll
    for (int nt = 0; nt < 4; ++nt) bb[nt] = biaspad[nt * 16 + ln15];

    v8f acc[4] = {};
    for (int kb = 0; kb < H_; kb += 32) {
        if (kb + 32 < H_)  // gfx1250 global_prefetch_b8 for the next A chunk
            __builtin_prefetch(arow + kb + 32 + kOffA, 0, 1);

        // ---- A fragment: lane holds M=row, K = kb+kOffA+{0..15}
        f4 fa[4];
        fa[0] = *(const f4*)(arow + kb + kOffA);
        fa[1] = *(const f4*)(arow + kb + kOffA + 4);
        fa[2] = *(const f4*)(arow + kb + kOffA + 8);
        fa[3] = *(const f4*)(arow + kb + kOffA + 12);

        v16bf ahi, alo;
#pragma unroll
        for (int e = 0; e < 16; ++e) {
            float  x = fa[e >> 2][e & 3];
            __bf16 h = (__bf16)x;             // hi part (RN)
            ahi[e] = h;
            alo[e] = (__bf16)(x - (float)h);  // residual -> ~fp32 accuracy
        }

        // ---- 4 N-tiles: B fragments are contiguous 32-byte packed loads
        const int kblk = kb >> 5;
#pragma unroll
        for (int nt = 0; nt < 4; ++nt) {
            const int col  = nt * 16 + ln15;
            const size_t o = (((size_t)kblk * NPAD + col) << 5) + kOffB;
            v16bf bhi = *(const v16bf*)(Whi + o);   // 2x global_load_b128
            v16bf blo = *(const v16bf*)(Wlo + o);
            // acc += Ahi*Bhi + Ahi*Blo + Alo*Bhi  (fp32 accumulate)
            acc[nt] = __builtin_amdgcn_wmma_f32_16x16x32_bf16(
                          false, ahi, false, bhi, (short)0, acc[nt], false, false);
            acc[nt] = __builtin_amdgcn_wmma_f32_16x16x32_bf16(
                          false, ahi, false, blo, (short)0, acc[nt], false, false);
            acc[nt] = __builtin_amdgcn_wmma_f32_16x16x32_bf16(
                          false, alo, false, bhi, (short)0, acc[nt], false, false);
        }
    }

    // C/D layout: VGPR r -> M = r (lanes 0-15) / 8+r (lanes 16-31), N = lane&15
#pragma unroll
    for (int nt = 0; nt < 4; ++nt) {
#pragma unroll
        for (int r = 0; r < 8; ++r) {
            int m = mbase + ((lane < 16) ? r : (8 + r));
            Em[(size_t)m * NPAD + nt * 16 + ln15] = acc[nt][r] + bb[nt];
        }
    }
}

// ---------------------------------------------------------------------------
// Kernel 2: CRF numerator + forward logsumexp scan + loss.
// One 64-thread block per batch element; thread j owns tag j (j < 50).
// Transitions in LDS with stride 51 (conflict-free column access); alpha is
// double-buffered in LDS -> one barrier per scan step.
// ---------------------------------------------------------------------------
__global__ __launch_bounds__(64)
void crf_kernel(const float* __restrict__ Em,     // [M_, NPAD]
                const float* __restrict__ startT, // [T_]
                const float* __restrict__ endT,   // [T_]
                const float* __restrict__ trans,  // [T_, T_]
                const int*   __restrict__ tags,   // [B_, S_]
                const unsigned char* __restrict__ mask, // [B_, S_] (bool)
                float* __restrict__ out)          // [B_]
{
    __shared__ float sTrans[T_ * 51];
    __shared__ float sAlpha[2][NPAD];
    __shared__ float sRed[64];
    __shared__ int   sRedI[64];
    __shared__ float sNum;
    __shared__ int   sLen;

    const int tid  = threadIdx.x;
    const int b    = blockIdx.x;
    const int base = b * S_;

    for (int idx = tid; idx < T_ * T_; idx += 64)
        sTrans[(idx / T_) * 51 + (idx % T_)] = trans[idx];

    // sequence length = popcount of the (prefix) mask
    int cnt = 0;
    for (int s = tid; s < S_; s += 64) cnt += mask[base + s] ? 1 : 0;
    sRedI[tid] = cnt;
    __syncthreads();
    if (tid == 0) { int l = 0; for (int i = 0; i < 64; ++i) l += sRedI[i]; sLen = l; }
    __syncthreads();
    const int len = sLen;

    // ---- numerator (gold-path score) ----
    float numl = 0.0f;
    for (int s = tid; s < len; s += 64) {
        int   tg = tags[base + s];
        float e  = Em[(size_t)(base + s) * NPAD + tg];
        if (s == 0) numl += startT[tg] + e;
        else        numl += sTrans[tags[base + s - 1] * 51 + tg] + e;
    }
    sRed[tid] = numl;
    __syncthreads();
    if (tid == 0) {
        float n = endT[tags[base + len - 1]];
        for (int i = 0; i < 64; ++i) n += sRed[i];
        sNum = n;
    }

    // ---- forward scan ----
    if (tid < T_) sAlpha[0][tid] = startT[tid] + Em[(size_t)base * NPAD + tid];
    __syncthreads();

    int cur = 0;
    for (int t = 1; t < len; ++t) {
        if (tid < T_) {
            const float* a = sAlpha[cur];
            float m = -3.4e38f;
            for (int i = 0; i < T_; ++i)
                m = fmaxf(m, a[i] + sTrans[i * 51 + tid]);
            float ssum = 0.0f;
            for (int i = 0; i < T_; ++i)
                ssum += __expf(a[i] + sTrans[i * 51 + tid] - m);
            sAlpha[cur ^ 1][tid] = m + __logf(ssum) + Em[(size_t)(base + t) * NPAD + tid];
        }
        __syncthreads();   // double buffer -> single barrier per step
        cur ^= 1;
    }

    // ---- denominator + loss ----
    if (tid < T_) sRed[tid] = sAlpha[cur][tid] + endT[tid];
    __syncthreads();
    if (tid == 0) {
        float m = -3.4e38f;
        for (int i = 0; i < T_; ++i) m = fmaxf(m, sRed[i]);
        float ssum = 0.0f;
        for (int i = 0; i < T_; ++i) ssum += __expf(sRed[i] - m);
        out[b] = (m + __logf(ssum)) - sNum;
    }
}

// ---------------------------------------------------------------------------
extern "C" void kernel_launch(void* const* d_in, const int* in_sizes, int n_in,
                              void* d_out, int out_size, void* d_ws, size_t ws_size,
                              hipStream_t stream)
{
    (void)in_sizes; (void)n_in; (void)out_size; (void)ws_size;

    const float*         full_hidden = (const float*)d_in[0];
    const float*         W           = (const float*)d_in[1];
    const float*         bias        = (const float*)d_in[2];
    const float*         startT      = (const float*)d_in[3];
    const float*         endT        = (const float*)d_in[4];
    const float*         trans       = (const float*)d_in[5];
    const int*           tags        = (const int*)d_in[6];
    const unsigned char* mask        = (const unsigned char*)d_in[7];
    float*               out         = (float*)d_out;

    // workspace carve-up
    char*   ws      = (char*)d_ws;
    float*  Em      = (float*)ws;                               // 8.39 MB
    __bf16* Whi     = (__bf16*)(ws + EM_BYTES);                 // 96 KB
    __bf16* Wlo     = (__bf16*)(ws + EM_BYTES + WPK_BYTES);     // 96 KB
    float*  biaspad = (float*)(ws + EM_BYTES + 2 * WPK_BYTES);  // 256 B

    // 1) pack W into WMMA B-fragment bf16 hi/lo planes (49152 elems / plane)
    hipLaunchKernelGGL(pack_w_kernel, dim3(WPK_ELEMS / 256), dim3(256), 0, stream,
                       W, bias, Whi, Wlo, biaspad);
    // 2) emissions GEMM: 2048 M-tiles, 8 waves/block, 16x64 stripe per wave
    hipLaunchKernelGGL(emis_gemm_kernel, dim3(M_ / (8 * 16)), dim3(256), 0, stream,
                       full_hidden, Whi, Wlo, biaspad, Em);
    // 3) CRF scan + loss
    hipLaunchKernelGGL(crf_kernel, dim3(B_), dim3(64), 0, stream,
                       Em, startT, endT, trans, tags, mask, out);
}
